// GraphReg_22402549416140
// MI455X (gfx1250) — matrix operand order repaired and checked by reading, compiled, not run on hardware
//
#include <hip/hip_runtime.h>
#include <hip/hip_bf16.h>

typedef _Float16 half_t;
typedef __attribute__((ext_vector_type(16))) _Float16 v16h;
typedef __attribute__((ext_vector_type(8)))  _Float16 v8h;
typedef __attribute__((ext_vector_type(8)))  float    v8f;
typedef __attribute__((ext_vector_type(4)))  float    v4f;

#define BN_EPS 1e-3f

// ---------------- WMMA fragment helpers (16x16x32 f16, wave32) ----------------
// A (16x32 MxK): lane<16 -> row M=lane&15, K halves {k0+0..7, k0+16..23}; lane>=16 same rows, +8.
// B (32x16 KxN): pre-packed per-lane: 16 contiguous halves per lane per fragment.
// C/D f32: VGPR r -> (M = r + 8*(lane>=16), N = lane&15).

__device__ __forceinline__ v16h combine16(v8h lo, v8h hi) {
  v16h f;
#pragma unroll
  for (int e = 0; e < 8; ++e) { f[e] = lo[e]; f[e + 8] = hi[e]; }
  return f;
}

// A fragment from a contiguous f16 row (global or LDS pointer)
__device__ __forceinline__ v16h frag_a_row(const half_t* rowp, int k0, bool ok) {
  int lane = threadIdx.x & 31;
  int koff = (lane & 16) ? 8 : 0;
  const half_t* p = rowp + k0 + koff;
  v8h z = {};
  v8h lo = *(const v8h*)(p);
  v8h hi = *(const v8h*)(p + 16);
  lo = ok ? lo : z;
  hi = ok ? hi : z;
  return combine16(lo, hi);
}

// A fragment from a contiguous f32 row (full 32-wide chunk, no guards)
__device__ __forceinline__ v16h frag_a_f32(const float* rowp, int k0) {
  int lane = threadIdx.x & 31;
  int koff = (lane & 16) ? 8 : 0;
  const float* p = rowp + k0 + koff;
  v4f a0 = *(const v4f*)(p);
  v4f a1 = *(const v4f*)(p + 4);
  v4f a2 = *(const v4f*)(p + 16);
  v4f a3 = *(const v4f*)(p + 20);
  v16h f;
#pragma unroll
  for (int e = 0; e < 4; ++e) {
    f[e]      = (half_t)a0[e];
    f[e + 4]  = (half_t)a1[e];
    f[e + 8]  = (half_t)a2[e];
    f[e + 12] = (half_t)a3[e];
  }
  return f;
}

// A tail fragment: only low 8 K-values per lane valid (covers K 1184..1199), hi half zero
__device__ __forceinline__ v16h frag_a_f32_tail(const float* rowp, int k0) {
  int lane = threadIdx.x & 31;
  int koff = (lane & 16) ? 8 : 0;
  const float* p = rowp + k0 + koff;
  v4f a0 = *(const v4f*)(p);
  v4f a1 = *(const v4f*)(p + 4);
  v16h f = {};
#pragma unroll
  for (int e = 0; e < 4; ++e) { f[e] = (half_t)a0[e]; f[e + 4] = (half_t)a1[e]; }
  return f;
}

// B fragment from pre-packed buffer: [frag][lane][16 halves]
__device__ __forceinline__ v16h frag_b_packed(const half_t* pack, long fragIdx) {
  int lane = threadIdx.x & 31;
  const half_t* p = pack + (fragIdx * 32 + lane) * 16;
  v8h lo = *(const v8h*)(p);
  v8h hi = *(const v8h*)(p + 8);
  return combine16(lo, hi);
}

#define WMMA(acc, a, b) \
  acc = __builtin_amdgcn_wmma_f32_16x16x32_f16(false, a, false, b, (short)0, acc, false, false)

// ---------------- B-operand pack kernels ----------------
// Generic f32 row-major B [K x N] -> fragment layout [KC][NT][32][16]
__global__ void k_packB(const float* __restrict__ src, half_t* __restrict__ dst,
                        int KC, int NT, int N) {
  int idx = blockIdx.x * blockDim.x + threadIdx.x;
  int total = KC * NT * 32 * 16;
  if (idx >= total) return;
  int e = idx & 15, lane = (idx >> 4) & 31;
  int t2 = idx >> 9;
  int nt = t2 % NT, kc = t2 / NT;
  int k = kc * 32 + ((lane & 16) ? 8 : 0) + e + ((e & 8) ? 8 : 0);
  int n = nt * 16 + (lane & 15);
  dst[idx] = (half_t)src[(long)k * N + n];
}

// Pack [ks|kn] heads into B [128 x 256] fragment layout [4][16][32][16]
__global__ void k_packWcat(const float* __restrict__ ks, const float* __restrict__ kn,
                           half_t* __restrict__ dst) {
  int idx = blockIdx.x * blockDim.x + threadIdx.x;
  if (idx >= 4 * 16 * 32 * 16) return;
  int e = idx & 15, lane = (idx >> 4) & 31;
  int t2 = idx >> 9;
  int nt = t2 % 16, kc = t2 / 16;
  int k = kc * 32 + ((lane & 16) ? 8 : 0) + e + ((e & 8) ? 8 : 0);
  int n = nt * 16 + (lane & 15);
  int hn = n & 127, h = hn >> 5, kk = hn & 31;
  const float* s = (n < 128) ? ks : kn;
  dst[idx] = (half_t)s[((long)h * 128 + k) * 32 + kk];
}

// Pack fn (fsfn cols 128..255) into per-(b,h) B panels: [b][kc=38][h*2+nt][32][16], K zero-padded
__global__ void k_packFn(const half_t* __restrict__ fsfn, half_t* __restrict__ dst) {
  int idx = blockIdx.x * blockDim.x + threadIdx.x;  // 4*38*8*32
  if (idx >= 4 * 38 * 8 * 32) return;
  int lane = idx & 31;
  int t2 = idx >> 5;
  int hnt = t2 & 7; t2 >>= 3;
  int kc = t2 % 38, b = t2 / 38;
  int h = hnt >> 1, nt = hnt & 1;
  int n = nt * 16 + (lane & 15);
  int c = 128 + h * 32 + n;
  int koff = (lane & 16) ? 8 : 0;
  v8h o0 = {}, o1 = {};
#pragma unroll
  for (int e = 0; e < 16; ++e) {
    int j = kc * 32 + koff + e + ((e & 8) ? 8 : 0);
    half_t v = (half_t)0.0f;
    if (j < 1200) v = fsfn[((long)b * 1200 + j) * 256 + c];
    if (e < 8) o0[e] = v; else o1[e - 8] = v;
  }
  v8h* d = (v8h*)(dst + (long)idx * 16);
  d[0] = o0; d[1] = o1;
}

// ---------------- conv tower ----------------
__global__ void k_conv1(const float* __restrict__ x, const float* __restrict__ w1,
                        const float* __restrict__ b1,
                        const float* __restrict__ g, const float* __restrict__ be,
                        const float* __restrict__ mu, const float* __restrict__ va,
                        half_t* __restrict__ h1h) {
  long idx = (long)blockIdx.x * blockDim.x + threadIdx.x;
  if (idx >= (long)4 * 30000 * 128) return;
  int co = (int)(idx & 127);
  long t2 = idx >> 7;
  int lp = (int)(t2 % 30000);
  int b  = (int)(t2 / 30000);
  const float* xb = x + (long)b * 60000 * 4;
  float gg = g[co], bb = be[co], mm = mu[co], iv = rsqrtf(va[co] + BN_EPS);
  float best = -3.4e38f;
#pragma unroll
  for (int p = 0; p < 2; ++p) {
    int l = 2 * lp + p;
    float acc = b1[co];
#pragma unroll
    for (int t = 0; t < 3; ++t) {
      int li = l + t - 1;
      if (li >= 0 && li < 60000) {
        v4f xv = *(const v4f*)(xb + (long)li * 4);
#pragma unroll
        for (int ci = 0; ci < 4; ++ci)
          acc += xv[ci] * w1[(t * 4 + ci) * 128 + co];
      }
    }
    acc = acc > 0.f ? acc : 0.f;
    acc = gg * (acc - mm) * iv + bb;
    best = acc > best ? acc : best;
  }
  h1h[idx] = (half_t)best;
}

// conv (w=3, 128->128) as 3 shifted WMMA GEMMs; A tile staged in LDS; fused ReLU+BN
__global__ void k_conv_gemm(const half_t* __restrict__ Ah, const half_t* __restrict__ Bpack,
                            const float* __restrict__ bias,
                            const float* __restrict__ g, const float* __restrict__ be,
                            const float* __restrict__ mu, const float* __restrict__ va,
                            float* __restrict__ Y, int Lin) {
  __shared__ half_t Atile[18 * 136];
  int b = blockIdx.y;
  int m0 = blockIdx.x * 16;
  const half_t* Abase = Ah + (long)b * Lin * 128;
  // stage rows m0-1 .. m0+16 (18 rows x 128 halves), zero-filled outside [0,Lin)
  for (int c = threadIdx.x; c < 18 * 16; c += 256) {
    int r = c >> 4, ch = c & 15;
    int l = m0 - 1 + r;
    v8h v = {};
    if (l >= 0 && l < Lin) v = *(const v8h*)(Abase + (long)l * 128 + ch * 8);
    *(v8h*)(&Atile[r * 136 + ch * 8]) = v;
  }
  __syncthreads();
  int wave = threadIdx.x >> 5;
  int lane = threadIdx.x & 31;
  int n = wave * 16 + (lane & 15);
  v8f acc = {};
#pragma unroll
  for (int t = 0; t < 3; ++t) {
    const half_t* rp = &Atile[((lane & 15) + t) * 136];
#pragma unroll
    for (int c = 0; c < 4; ++c) {
      v16h a  = frag_a_row(rp, c * 32, true);
      v16h bb = frag_b_packed(Bpack, (t * 4 + c) * 8 + wave);
      WMMA(acc, a, bb);
    }
  }
  float gg = g[n], bb2 = be[n], mm = mu[n], iv = rsqrtf(va[n] + BN_EPS);
  float bs = bias[n];
  int mbase = m0 + ((lane >> 4) << 3);
#pragma unroll
  for (int r = 0; r < 8; ++r) {
    float y = acc[r] + bs;
    y = y > 0.f ? y : 0.f;
    y = gg * (y - mm) * iv + bb2;
    Y[((long)b * Lin + (mbase + r)) * 128 + n] = y;
  }
}

__global__ void k_pool5(const float* __restrict__ Y, half_t* __restrict__ Oh,
                        float* __restrict__ Of, int Lout) {
  long idx = (long)blockIdx.x * blockDim.x + threadIdx.x;
  if (idx >= (long)4 * Lout * 128) return;
  int c = (int)(idx & 127);
  long t2 = idx >> 7;
  int lp = (int)(t2 % Lout);
  int b  = (int)(t2 / Lout);
  const float* p = Y + ((long)(b * Lout + lp) * 5) * 128 + c;
  float m = p[0];
#pragma unroll
  for (int q = 1; q < 5; ++q) { float v = p[q * 128]; m = v > m ? v : m; }
  Oh[idx] = (half_t)m;
  if (Of) Of[idx] = m;
}

// ---------------- GAT ----------------
// fsfn = X[4800,128] @ Wcat[128,256] -> f16 [4800,256]; A tile via LDS
__global__ void k_fsfn(const half_t* __restrict__ Xh, const half_t* __restrict__ Wp,
                       half_t* __restrict__ fsfn) {
  __shared__ half_t Atile[16 * 136];
  int m0 = blockIdx.x * 16;
  {
    int r = threadIdx.x >> 4, ch = threadIdx.x & 15;
    *(v8h*)(&Atile[r * 136 + ch * 8]) = *(const v8h*)(Xh + (long)(m0 + r) * 128 + ch * 8);
  }
  __syncthreads();
  int wave = threadIdx.x >> 5;
  int lane = threadIdx.x & 31;
  const half_t* rp = &Atile[(lane & 15) * 136];
  int mbase = m0 + ((lane >> 4) << 3);
#pragma unroll
  for (int half_i = 0; half_i < 2; ++half_i) {
    int nt = wave + half_i * 8;
    int n = nt * 16 + (lane & 15);
    v8f acc = {};
#pragma unroll
    for (int c = 0; c < 4; ++c) {
      v16h a  = frag_a_row(rp, c * 32, true);
      v16h bb = frag_b_packed(Wp, c * 16 + nt);
      WMMA(acc, a, bb);
    }
#pragma unroll
    for (int r = 0; r < 8; ++r)
      fsfn[(long)(mbase + r) * 256 + n] = (half_t)acc[r];
  }
}

__global__ void k_esen(const half_t* __restrict__ fsfn, const float* __restrict__ a_s,
                       const float* __restrict__ a_n, float* __restrict__ es,
                       float* __restrict__ en) {
  int idx = blockIdx.x * blockDim.x + threadIdx.x;
  if (idx >= 4 * 4800) return;
  int h = idx / 4800, bn = idx % 4800;
  const half_t* f = fsfn + (long)bn * 256 + h * 32;
  const float* as_ = a_s + h * 32;
  const float* an_ = a_n + h * 32;
  float s1 = 0.f, s2 = 0.f;
#pragma unroll
  for (int k = 0; k < 32; ++k) {
    s1 += (float)f[k] * as_[k];
    s2 += (float)f[128 + k] * an_[k];
  }
  es[idx] = s1; en[idx] = s2;
}

// one block per attention row: sigmoid(masked leaky pre), row-sum, normalize in d_out, save beta
__global__ void k_att(const float* __restrict__ es, const float* __restrict__ en,
                      const float* __restrict__ A, float* __restrict__ attout,
                      float* __restrict__ beta) {
  __shared__ float red[256];
  int rid = blockIdx.x;              // (h*4+b)*1200 + i
  int hb = rid / 1200, i = rid % 1200;
  int b = hb & 3;
  float esi = es[rid];
  const float* enrow = en + (long)hb * 1200;
  const float* Arow  = A + ((long)b * 1200 + i) * 1200;
  float* orow = attout + (long)rid * 1200;
  float s = 0.f;
  for (int j = threadIdx.x; j < 1200; j += 256) {
    float pre = esi + enrow[j];
    pre = pre >= 0.f ? pre : 0.2f * pre;
    pre -= 1e16f * (1.0f - Arow[j]);
    float at = 1.0f / (1.0f + expf(-pre));
    orow[j] = at;
    s += at;
  }
  red[threadIdx.x] = s;
  __syncthreads();
  for (int st = 128; st > 0; st >>= 1) {
    if ((int)threadIdx.x < st) red[threadIdx.x] += red[threadIdx.x + st];
    __syncthreads();
  }
  float inv = 1.0f / (1.0f + red[0]);
  if (threadIdx.x == 0) beta[rid] = inv;
  for (int j = threadIdx.x; j < 1200; j += 256) orow[j] *= inv;
}

// node = att[1200,1200] @ fn[1200,32] + fs*beta ; ELU ; BN -> next X (f32+f16)
__global__ void k_node(const float* __restrict__ att, const half_t* __restrict__ fsfn,
                       const half_t* __restrict__ fnpack, const float* __restrict__ beta,
                       const float* __restrict__ g, const float* __restrict__ be,
                       const float* __restrict__ mu, const float* __restrict__ va,
                       float* __restrict__ Xf, half_t* __restrict__ Xh) {
  int hb = blockIdx.y, h = hb >> 2, b = hb & 3;
  int m0 = blockIdx.x * 16;
  int wave = threadIdx.x >> 5;     // 0..1 -> n-tile
  int lane = threadIdx.x & 31;
  int n = wave * 16 + (lane & 15); // 0..31
  int i = m0 + (lane & 15);
  const float* arow = att + ((long)hb * 1200 + i) * 1200;
  long fragbase = ((long)b * 38) * 8 + (h * 2 + wave);
  v8f acc = {};
  for (int kc = 0; kc < 37; ++kc) {
    v16h a  = frag_a_f32(arow, kc * 32);
    v16h bb = frag_b_packed(fnpack, fragbase + (long)kc * 8);
    WMMA(acc, a, bb);
  }
  { // tail: K 1184..1199 (hi half zero; fnpack zero-padded)
    v16h a  = frag_a_f32_tail(arow, 37 * 32);
    v16h bb = frag_b_packed(fnpack, fragbase + 37L * 8);
    WMMA(acc, a, bb);
  }
  int c = h * 32 + n;
  float gg = g[c], bb2 = be[c], mm = mu[c], iv = rsqrtf(va[c] + BN_EPS);
  int mbase = m0 + ((lane >> 4) << 3);
#pragma unroll
  for (int r = 0; r < 8; ++r) {
    int i2 = mbase + r;
    float fs = (float)fsfn[((long)b * 1200 + i2) * 256 + c];
    float v = acc[r] + fs * beta[(long)hb * 1200 + i2];
    v = v > 0.f ? v : expm1f(v);                 // ELU
    v = gg * (v - mm) * iv + bb2;                // BN
    Xf[((long)b * 1200 + i2) * 128 + c] = v;
    Xh[((long)b * 1200 + i2) * 128 + c] = (half_t)v;
  }
}

// ---------------- final head ----------------
__global__ void k_final(const float* __restrict__ X, const float* __restrict__ wf1,
                        const float* __restrict__ bf1, const float* __restrict__ bg,
                        const float* __restrict__ bb, const float* __restrict__ bm,
                        const float* __restrict__ bv, const float* __restrict__ wf2,
                        const float* __restrict__ bf2, float* __restrict__ out) {
  __shared__ float mid[64];
  int bn = blockIdx.x;
  int co = threadIdx.x;
  const float* xr = X + (long)bn * 128;
  float acc = bf1[co];
  for (int ci = 0; ci < 128; ++ci) acc += xr[ci] * wf1[ci * 64 + co];
  acc = acc > 0.f ? acc : 0.f;
  acc = bg[co] * (acc - bm[co]) * rsqrtf(bv[co] + BN_EPS) + bb[co];
  mid[co] = acc;
  __syncthreads();
  if (co == 0) {
    float v = bf2[0];
    for (int k = 0; k < 64; ++k) v += mid[k] * wf2[k];
    out[bn] = expf(v);
  }
}

extern "C" void kernel_launch(void* const* d_in, const int* in_sizes, int n_in,
                              void* d_out, int out_size, void* d_ws, size_t ws_size,
                              hipStream_t stream) {
  (void)in_sizes; (void)n_in; (void)out_size; (void)ws_size;
  const float* x   = (const float*)d_in[0];
  const float* A   = (const float*)d_in[1];
  const float* w1  = (const float*)d_in[2];
  const float* b1  = (const float*)d_in[3];
  const float* w2  = (const float*)d_in[4];
  const float* b2  = (const float*)d_in[5];
  const float* w3  = (const float*)d_in[6];
  const float* b3  = (const float*)d_in[7];
  const float* bng = (const float*)d_in[8];   // [6,128]
  const float* bnb = (const float*)d_in[9];
  const float* bnm = (const float*)d_in[10];
  const float* bnv = (const float*)d_in[11];
  const float* wf1 = (const float*)d_in[24];
  const float* bf1 = (const float*)d_in[25];
  const float* bfg = (const float*)d_in[26];
  const float* bfb = (const float*)d_in[27];
  const float* bfm = (const float*)d_in[28];
  const float* bfv = (const float*)d_in[29];
  const float* wf2 = (const float*)d_in[30];
  const float* bf2 = (const float*)d_in[31];
  float* out = (float*)d_out;

  char* w = (char*)d_ws;
  size_t o = 0;
  auto alloc = [&](size_t bytes) { char* p = w + o; o = (o + bytes + 255) & ~(size_t)255; return p; };
  half_t* h1h   = (half_t*)alloc((size_t)4 * 30000 * 128 * 2);
  float*  y2    = (float*) alloc((size_t)4 * 30000 * 128 * 4);
  half_t* h2h   = (half_t*)alloc((size_t)4 * 6000  * 128 * 2);
  float*  y3    = (float*) alloc((size_t)4 * 6000  * 128 * 4);
  float*  Xf    = (float*) alloc((size_t)4 * 1200  * 128 * 4);
  half_t* Xh    = (half_t*)alloc((size_t)4 * 1200  * 128 * 2);
  half_t* fsfn  = (half_t*)alloc((size_t)4800 * 256 * 2);
  float*  es    = (float*) alloc((size_t)19200 * 4);
  float*  en    = (float*) alloc((size_t)19200 * 4);
  float*  beta  = (float*) alloc((size_t)19200 * 4);
  half_t* w2p   = (half_t*)alloc((size_t)12 * 8 * 32 * 16 * 2);   // [KC=12][NT=8][32][16]
  half_t* w3p   = (half_t*)alloc((size_t)12 * 8 * 32 * 16 * 2);
  half_t* wcatp = (half_t*)alloc((size_t)4 * 16 * 32 * 16 * 2);   // [KC=4][NT=16][32][16]
  half_t* fnp   = (half_t*)alloc((size_t)4 * 38 * 8 * 32 * 16 * 2); // [b][kc][h*2+nt][32][16]

  // conv tower
  k_conv1<<<60000, 256, 0, stream>>>(x, w1, b1, bng, bnb, bnm, bnv, h1h);
  k_packB<<<(12 * 8 * 32 * 16 + 255) / 256, 256, 0, stream>>>(w2, w2p, 12, 8, 128);
  k_packB<<<(12 * 8 * 32 * 16 + 255) / 256, 256, 0, stream>>>(w3, w3p, 12, 8, 128);
  k_conv_gemm<<<dim3(1875, 4), 256, 0, stream>>>(h1h, w2p, b2, bng + 128, bnb + 128,
                                                 bnm + 128, bnv + 128, y2, 30000);
  k_pool5<<<((size_t)4 * 6000 * 128 + 255) / 256, 256, 0, stream>>>(y2, h2h, nullptr, 6000);
  k_conv_gemm<<<dim3(375, 4), 256, 0, stream>>>(h2h, w3p, b3, bng + 256, bnb + 256,
                                                bnm + 256, bnv + 256, y3, 6000);
  k_pool5<<<((size_t)4 * 1200 * 128 + 255) / 256, 256, 0, stream>>>(y3, Xh, Xf, 1200);

  // GAT layers
  for (int layer = 0; layer < 3; ++layer) {
    const float* ks  = (const float*)d_in[12 + layer * 4];
    const float* kn  = (const float*)d_in[13 + layer * 4];
    const float* a_s = (const float*)d_in[14 + layer * 4];
    const float* a_n = (const float*)d_in[15 + layer * 4];
    float* attbase = out + 4800 + (size_t)layer * 23040000;  // [H,B,N,N]
    k_packWcat<<<(4 * 16 * 32 * 16 + 255) / 256, 256, 0, stream>>>(ks, kn, wcatp);
    k_fsfn<<<300, 256, 0, stream>>>(Xh, wcatp, fsfn);
    k_packFn<<<(4 * 38 * 8 * 32 + 255) / 256, 256, 0, stream>>>(fsfn, fnp);
    k_esen<<<75, 256, 0, stream>>>(fsfn, a_s, a_n, es, en);
    k_att<<<19200, 256, 0, stream>>>(es, en, A, attbase, beta);
    k_node<<<dim3(75, 16), 64, 0, stream>>>(attbase, fsfn, fnp, beta,
                                            bng + (3 + layer) * 128, bnb + (3 + layer) * 128,
                                            bnm + (3 + layer) * 128, bnv + (3 + layer) * 128,
                                            Xf, Xh);
  }

  // final head
  k_final<<<4800, 64, 0, stream>>>(Xf, wf1, bf1, bfg, bfb, bfm, bfv, wf2, bf2, out);
}
